// Attention_49761491091709
// MI455X (gfx1250) — matrix operand order repaired and checked by reading, not compiled, not run on hardware
//
#include <hip/hip_runtime.h>
#include <hip/hip_bf16.h>

#define BATCH 2
#define CCH   128
#define NSP   4096            // 16*16*16
#define NGRP  32
#define CPG   (CCH / NGRP)    // 4
#define GN_EPS 1e-5f

typedef __attribute__((ext_vector_type(16))) __bf16 v16bf;
typedef __attribute__((ext_vector_type(8)))  __bf16 v8bf;
typedef __attribute__((ext_vector_type(8)))  float  v8f;
typedef __attribute__((ext_vector_type(2)))  float  v2f;

__device__ inline v8bf cvt8(v8f a) {
    v8bf r;
    for (int i = 0; i < 8; ++i) r[i] = (__bf16)a[i];
    return r;
}

// LDS byte offset of a __shared__ object: LDS aperture keeps the offset in
// addr[31:0] of the generic pointer (ISA 10.2 aperture mapping).
__device__ inline uint32_t ldsoff(const void* p) {
    return (uint32_t)(uintptr_t)p;
}

// ASYNCcnt-tracked 16B global->LDS copy (CDNA5 async memory path).
__device__ inline void async_ld16(uint32_t lds_off, const void* gaddr) {
    asm volatile("global_load_async_to_lds_b128 %0, %1, off"
                 :: "v"(lds_off), "v"(gaddr) : "memory");
}
__device__ inline void wait_async(int n) {
    if (n == 0) asm volatile("s_wait_asynccnt 0x0" ::: "memory");
    else        asm volatile("s_wait_asynccnt 0x8" ::: "memory");
}

// ---------------------------------------------------------------------------
// Kernel 1: q/k/v projections, full-f32 WMMA (16x16x4), bias fused.
// One block = one (tensor, batch, 16-col n-tile); 8 waves cover all 128 output
// channels and share one async-staged 8 KB input tile in LDS.
// q,k written [B][N][C] bf16; v written [B][C][N] bf16.
// ---------------------------------------------------------------------------
__global__ __launch_bounds__(256) void qkv_proj_kernel(
    const float* __restrict__ query, const float* __restrict__ key_,
    const float* __restrict__ value,
    const float* __restrict__ wq, const float* __restrict__ bq,
    const float* __restrict__ wk, const float* __restrict__ bk,
    const float* __restrict__ wv, const float* __restrict__ bv,
    __bf16* __restrict__ qbf, __bf16* __restrict__ kbf, __bf16* __restrict__ vbf)
{
    __shared__ __attribute__((aligned(64))) float xs[CCH][16];   // 8 KB input tile

    const int tid  = threadIdx.x;
    const int lane = tid & 31;
    const int wave = tid >> 5;
    const int hi   = lane >> 4;
    const int l16  = lane & 15;

    const int blk = blockIdx.x;                 // 3 * B * 256 = 1536
    const int t   = blk / (BATCH * 256);        // 0=q 1=k 2=v
    int rem       = blk % (BATCH * 256);
    const int b   = rem / 256;
    const int n0  = (rem % 256) * 16;
    const int o0  = wave * 16;

    const float* in   = (t == 0) ? query : (t == 1) ? key_ : value;
    const float* W    = (t == 0) ? wq    : (t == 1) ? wk   : wv;
    const float* bias = (t == 0) ? bq    : (t == 1) ? bk   : bv;
    in += (size_t)b * CCH * NSP;

    // async-stage input tile X[c][n0..n0+15]: 512 x 16B chunks, 2 per thread
    for (int p = 0; p < 2; ++p) {
        const int idx = p * 256 + tid;
        const int c   = idx >> 2;
        const int col = (idx & 3) * 4;          // float elements
        async_ld16(ldsoff(&xs[c][col]), in + (size_t)c * NSP + n0 + col);
    }
    wait_async(0);
    __syncthreads();

    v8f acc = {0.f, 0.f, 0.f, 0.f, 0.f, 0.f, 0.f, 0.f};
    const int m = l16;                          // A row
    const int n = n0 + l16;                     // B/D col

    for (int kc = 0; kc < CCH; kc += 4) {
        // A fragment (16x4 f32): lane row m; K pair {kc+2hi, kc+2hi+1}
        v2f a = *(const v2f*)(W + (size_t)(o0 + m) * CCH + kc + 2 * hi);
        // B fragment (4x16 f32) from LDS tile
        v2f bb;
        bb[0] = xs[kc + 2 * hi][l16];
        bb[1] = xs[kc + 2 * hi + 1][l16];
        acc = __builtin_amdgcn_wmma_f32_16x16x4_f32(
            false, a, false, bb, (short)0, acc, false, false);
    }
    for (int r = 0; r < 8; ++r) acc[r] += bias[o0 + 8 * hi + r];

    if (t < 2) {
        __bf16* outp = (t == 0 ? qbf : kbf) + (size_t)b * NSP * CCH;
        *(v8bf*)(outp + (size_t)n * CCH + o0 + 8 * hi) = cvt8(acc);
    } else {
        __bf16* outp = vbf + (size_t)b * CCH * NSP;
        for (int r = 0; r < 8; ++r)
            outp[(size_t)(o0 + 8 * hi + r) * NSP + n] = (__bf16)acc[r];
    }
}

// ---------------------------------------------------------------------------
// Kernel 2: fused flash attention, double-buffered async K/V staging.
// One wave = one 16-query tile; 4 waves/block share each 32-key K/V block.
// S = Q K^T (bf16 WMMA, 4 K-steps), online softmax, O += P V (bf16 WMMA).
// ---------------------------------------------------------------------------
__device__ inline void issue_kv(const __bf16* kb, const __bf16* vb, int jb,
                                __bf16 (*ksh)[CCH], __bf16 (*vsh)[32], int tid)
{
    // K block: 32 rows x 256B = 512 chunks; V block: 128 rows x 64B = 512 chunks
    for (int p = 0; p < 4; ++p) {
        const int idx = p * 128 + tid;
        const int row = idx >> 4, col = (idx & 15) * 8;
        async_ld16(ldsoff(&ksh[row][col]), kb + (size_t)(jb + row) * CCH + col);
    }
    for (int p = 0; p < 4; ++p) {
        const int idx = p * 128 + tid;
        const int ch = idx >> 2, col = (idx & 3) * 8;
        async_ld16(ldsoff(&vsh[ch][col]), vb + (size_t)ch * NSP + jb + col);
    }
}

__global__ __launch_bounds__(128) void attn_kernel(
    const __bf16* __restrict__ qbf, const __bf16* __restrict__ kbf,
    const __bf16* __restrict__ vbf, __bf16* __restrict__ hbf)
{
    __shared__ __attribute__((aligned(64))) __bf16 kshm[2][32][CCH]; // 2 x 8 KB
    __shared__ __attribute__((aligned(64))) __bf16 vshm[2][CCH][32]; // 2 x 8 KB
    __shared__ __attribute__((aligned(64))) __bf16 pshm[4][16][32];  // 4 KB

    const int tid  = threadIdx.x;
    const int lane = tid & 31;
    const int wave = tid >> 5;
    const int hi   = lane >> 4;
    const int l16  = lane & 15;

    const int tile = blockIdx.x * 4 + wave;        // B * (N/16) = 512 tiles
    const int b    = tile / (NSP / 16);
    const int i0   = (tile % (NSP / 16)) * 16;

    const __bf16* qb = qbf + (size_t)b * NSP * CCH;
    const __bf16* kb = kbf + (size_t)b * NSP * CCH;
    const __bf16* vb = vbf + (size_t)b * CCH * NSP;

    // Q A-fragments (resident): 4 steps of K(=channel)=32
    v16bf qa[4];
    for (int s = 0; s < 4; ++s) {
        const int cb = 32 * s;
        union { v16bf v; v8bf h[2]; } u;
        u.h[0] = *(const v8bf*)(qb + (size_t)(i0 + l16) * CCH + cb + 16 * hi);
        u.h[1] = *(const v8bf*)(qb + (size_t)(i0 + l16) * CCH + cb + 16 * hi + 8);
        qa[s] = u.v;
    }

    v8f oacc[8];
    for (int t = 0; t < 8; ++t)
        for (int r = 0; r < 8; ++r) oacc[t][r] = 0.f;
    float mrow[8], lrow[8];
    for (int r = 0; r < 8; ++r) { mrow[r] = -1e30f; lrow[r] = 0.f; }

    issue_kv(kb, vb, 0, kshm[0], vshm[0], tid);    // prologue fill

    for (int jb = 0; jb < NSP; jb += 32) {
        const int  cur       = (jb >> 5) & 1;
        const bool have_next = (jb + 32 < NSP);
        if (have_next)                              // pipeline: fill other buffer
            issue_kv(kb, vb, jb + 32, kshm[cur ^ 1], vshm[cur ^ 1], tid);
        wait_async(have_next ? 8 : 0);              // in-order: current block landed
        __syncthreads();

        // --- S tiles: 16 queries x 32 keys (two 16x16 subtiles) ---
        v8f s0, s1;
        for (int r = 0; r < 8; ++r) { s0[r] = 0.f; s1[r] = 0.f; }
        for (int s = 0; s < 4; ++s) {
            const int cb = 32 * s;
            v16bf kf0 = *(const v16bf*)(&kshm[cur][l16][cb + 16 * hi]);
            s0 = __builtin_amdgcn_wmma_f32_16x16x32_bf16(
                false, qa[s], false, kf0, (short)0, s0, false, false);
            v16bf kf1 = *(const v16bf*)(&kshm[cur][16 + l16][cb + 16 * hi]);
            s1 = __builtin_amdgcn_wmma_f32_16x16x32_bf16(
                false, qa[s], false, kf1, (short)0, s1, false, false);
        }

        // --- online softmax (row in VGPR index, col across 16 lanes) ---
        float scale_r[8];
        for (int r = 0; r < 8; ++r) {
            float mx = fmaxf(s0[r], s1[r]);
            for (int msk = 8; msk >= 1; msk >>= 1)
                mx = fmaxf(mx, __shfl_xor(mx, msk, 32));
            const float mn = fmaxf(mrow[r], mx);
            const float sc = __expf(mrow[r] - mn);
            const float p0 = __expf(s0[r] - mn);
            const float p1 = __expf(s1[r] - mn);
            s0[r] = p0; s1[r] = p1;
            float rs = p0 + p1;
            for (int msk = 8; msk >= 1; msk >>= 1)
                rs += __shfl_xor(rs, msk, 32);
            lrow[r] = lrow[r] * sc + rs;
            mrow[r] = mn;
            scale_r[r] = sc;
        }
        for (int t = 0; t < 8; ++t)
            for (int r = 0; r < 8; ++r) oacc[t][r] *= scale_r[r];

        // --- transpose P (C/D layout -> A layout) through per-wave LDS ---
        for (int r = 0; r < 8; ++r) {
            pshm[wave][r + 8 * hi][l16]      = (__bf16)s0[r];
            pshm[wave][r + 8 * hi][16 + l16] = (__bf16)s1[r];
        }
        union { v16bf v; v8bf h[2]; } pu;    // wave-lockstep: DScnt orders st->ld
        pu.h[0] = *(const v8bf*)(&pshm[wave][l16][16 * hi]);
        pu.h[1] = *(const v8bf*)(&pshm[wave][l16][16 * hi + 8]);
        const v16bf pa = pu.v;

        // --- O += P V : 8 channel tiles of 16, K-dim = 32 keys ---
        for (int t = 0; t < 8; ++t) {
            v16bf vf = *(const v16bf*)(&vshm[cur][16 * t + l16][16 * hi]);
            oacc[t] = __builtin_amdgcn_wmma_f32_16x16x32_bf16(
                false, pa, false, vf, (short)0, oacc[t], false, false);
        }
        __syncthreads();                     // release buffer for overwrite
    }

    // normalize and store h as [B][N][C] bf16 (B-layout for the output proj)
    __bf16* hb = hbf + (size_t)b * NSP * CCH;
    for (int r = 0; r < 8; ++r) {
        const float inv = 1.0f / lrow[r];
        const int i = i0 + r + 8 * hi;
        for (int t = 0; t < 8; ++t)
            hb[(size_t)i * CCH + 16 * t + l16] = (__bf16)(oacc[t][r] * inv);
    }
}

// ---------------------------------------------------------------------------
// Kernel 3: output projection (bf16 WMMA) + bias + residual -> x (f32 [B][C][N])
// One block = one (batch, n-tile); 8 waves cover all 128 output channels and
// share one async-staged 4 KB h tile.
// ---------------------------------------------------------------------------
__global__ __launch_bounds__(256) void oproj_kernel(
    const __bf16* __restrict__ hbf, const float* __restrict__ wo,
    const float* __restrict__ bo, const float* __restrict__ value,
    float* __restrict__ x)
{
    __shared__ __attribute__((aligned(64))) __bf16 hs[16][CCH];  // 4 KB

    const int tid  = threadIdx.x;
    const int lane = tid & 31;
    const int wave = tid >> 5;
    const int hi   = lane >> 4;
    const int l16  = lane & 15;

    const int blk = blockIdx.x;              // B * 256 = 512
    const int b   = blk / 256;
    const int n0  = (blk % 256) * 16;
    const int o0  = wave * 16;

    const __bf16* hb = hbf + (size_t)b * NSP * CCH;

    // stage h tile [n0..n0+15][0..127]: 256 x 16B chunks, one per thread
    {
        const int row = tid >> 4, col = (tid & 15) * 8;
        async_ld16(ldsoff(&hs[row][col]), hb + (size_t)(n0 + row) * CCH + col);
    }
    wait_async(0);
    __syncthreads();

    v8f acc;
    for (int r = 0; r < 8; ++r) acc[r] = 0.f;

    for (int s = 0; s < 4; ++s) {
        const int cb = 32 * s;
        union { v16bf v; v8bf h[2]; } au;
        v8f w0 = *(const v8f*)(wo + (size_t)(o0 + l16) * CCH + cb + 16 * hi);
        v8f w1 = *(const v8f*)(wo + (size_t)(o0 + l16) * CCH + cb + 16 * hi + 8);
        au.h[0] = cvt8(w0);
        au.h[1] = cvt8(w1);
        v16bf bf = *(const v16bf*)(&hs[l16][cb + 16 * hi]);
        acc = __builtin_amdgcn_wmma_f32_16x16x32_bf16(
            false, au.v, false, bf, (short)0, acc, false, false);
    }

    const int n = n0 + l16;
    for (int r = 0; r < 8; ++r) {
        const int o = o0 + r + 8 * hi;
        const size_t idx = ((size_t)b * CCH + o) * NSP + n;
        x[idx] = acc[r] + bo[o] + value[idx];
    }
}

// ---------------------------------------------------------------------------
// Kernel 4: GroupNorm statistics — one block per (batch, group)
// ---------------------------------------------------------------------------
__global__ __launch_bounds__(256) void gn_stats_kernel(
    const float* __restrict__ x, float* __restrict__ stats)
{
    const int bg = blockIdx.x;                    // B*G = 64
    const int b  = bg / NGRP;
    const int g  = bg % NGRP;
    const float* xp = x + ((size_t)b * CCH + g * CPG) * NSP;
    const int total = CPG * NSP;                  // 16384

    float s = 0.f, ss = 0.f;
    for (int i = threadIdx.x; i < total; i += 256) {
        const float v = xp[i];
        s += v; ss += v * v;
    }
    __shared__ float sh1[256], sh2[256];
    sh1[threadIdx.x] = s; sh2[threadIdx.x] = ss;
    __syncthreads();
    for (int st = 128; st > 0; st >>= 1) {
        if (threadIdx.x < st) {
            sh1[threadIdx.x] += sh1[threadIdx.x + st];
            sh2[threadIdx.x] += sh2[threadIdx.x + st];
        }
        __syncthreads();
    }
    if (threadIdx.x == 0) {
        const float mean = sh1[0] / (float)total;
        const float var  = sh2[0] / (float)total - mean * mean;
        stats[2 * bg]     = mean;
        stats[2 * bg + 1] = rsqrtf(var + GN_EPS);
    }
}

// ---------------------------------------------------------------------------
// Kernel 5: fused GroupNorm + SiLU: out = y * sigmoid(y), y = gn(x)
// ---------------------------------------------------------------------------
__global__ __launch_bounds__(256) void gn_silu_kernel(
    const float* __restrict__ x, const float* __restrict__ stats,
    const float* __restrict__ gamma, const float* __restrict__ beta,
    float* __restrict__ out)
{
    const size_t base = ((size_t)blockIdx.x * 256 + threadIdx.x) * 4;
    const int c = (int)((base / NSP) % CCH);
    const int b = (int)(base / ((size_t)CCH * NSP));
    const int g = c / CPG;
    const float mean = stats[2 * (b * NGRP + g)];
    const float rstd = stats[2 * (b * NGRP + g) + 1];
    const float gm = gamma[c], bt = beta[c];

    const float4 xv = *(const float4*)(x + base);
    float4 ov;
    {
        float y;
        y = (xv.x - mean) * rstd * gm + bt; ov.x = y / (1.f + __expf(-y));
        y = (xv.y - mean) * rstd * gm + bt; ov.y = y / (1.f + __expf(-y));
        y = (xv.z - mean) * rstd * gm + bt; ov.z = y / (1.f + __expf(-y));
        y = (xv.w - mean) * rstd * gm + bt; ov.w = y / (1.f + __expf(-y));
    }
    *(float4*)(out + base) = ov;
}

// ---------------------------------------------------------------------------
extern "C" void kernel_launch(void* const* d_in, const int* in_sizes, int n_in,
                              void* d_out, int out_size, void* d_ws, size_t ws_size,
                              hipStream_t stream)
{
    const float* query = (const float*)d_in[0];
    const float* key_  = (const float*)d_in[1];
    const float* value = (const float*)d_in[2];
    const float* wq = (const float*)d_in[3];
    const float* bq = (const float*)d_in[4];
    const float* wk = (const float*)d_in[5];
    const float* bk = (const float*)d_in[6];
    const float* wv = (const float*)d_in[7];
    const float* bv = (const float*)d_in[8];
    const float* wo = (const float*)d_in[9];
    const float* bo = (const float*)d_in[10];
    const float* gamma = (const float*)d_in[11];
    const float* beta  = (const float*)d_in[12];
    float* out = (float*)d_out;

    char* ws = (char*)d_ws;
    const size_t MB = 1u << 20;
    __bf16* qbf = (__bf16*)(ws + 0 * MB);   // [B][N][C] bf16, 2 MB
    __bf16* kbf = (__bf16*)(ws + 2 * MB);   // [B][N][C] bf16, 2 MB
    __bf16* vbf = (__bf16*)(ws + 4 * MB);   // [B][C][N] bf16, 2 MB
    __bf16* hbf = (__bf16*)(ws + 6 * MB);   // [B][N][C] bf16, 2 MB
    float*  xb  = (float*) (ws + 8 * MB);   // [B][C][N] f32,  4 MB
    float*  st  = (float*) (ws + 12 * MB);  // 2*B*G floats

    qkv_proj_kernel<<<1536, 256, 0, stream>>>(query, key_, value,
                                              wq, bq, wk, bk, wv, bv,
                                              qbf, kbf, vbf);
    attn_kernel<<<128, 128, 0, stream>>>(qbf, kbf, vbf, hbf);
    oproj_kernel<<<512, 256, 0, stream>>>(hbf, wo, bo, value, xb);
    gn_stats_kernel<<<64, 256, 0, stream>>>(xb, st);
    gn_silu_kernel<<<1024, 256, 0, stream>>>(xb, st, gamma, beta, out);
}
